// SeqRnnDecoder_19679540150832
// MI455X (gfx1250) — compile-verified, matrix operand order checked by reading, not run on hardware
//
#include <hip/hip_runtime.h>
#include <hip/hip_bf16.h>

// ---------------------------------------------------------------------------
// Types for CDNA5 WMMA (wave32): 16x16x32 bf16 -> f32 accumulate
// ---------------------------------------------------------------------------
typedef __attribute__((ext_vector_type(16))) __bf16        v16bf;
typedef __attribute__((ext_vector_type(8)))  float         v8f;
typedef __attribute__((ext_vector_type(8)))  unsigned int  v8u;

#define B_   16
#define S_   512
#define EMB_ 512
#define HID_ 1024
#define VOC_ 32000
#define M_   (B_ * S_)   // 8192

__device__ __forceinline__ v8f wmma_bf16(v16bf a, v16bf b, v8f c) {
    // (neg_a, A, neg_b, B, c_mod, C, reuse_a, reuse_b)
    return __builtin_amdgcn_wmma_f32_16x16x32_bf16(false, a, false, b,
                                                   (short)0, c, false, false);
}

// Load one lane's A (or B) fragment of a 16x32 bf16 tile from a row-major
// [rows][ldk] bf16 matrix (global).  ISA layout: lane half h holds
// K = {0..7,16..23}+8h; element pairs are K-adjacent -> 8 dword loads
// (compiler merges into 2x global_load_b128).
__device__ __forceinline__ v16bf load_tile(const __bf16* __restrict__ base,
                                           size_t row, int ldk, int k0, int half) {
    const unsigned int* p =
        (const unsigned int*)(base + row * (size_t)ldk + (size_t)(k0 + 8 * half));
    v8u u;
    u[0] = p[0];  u[1] = p[1];  u[2] = p[2];  u[3] = p[3];   // K = 0,2,4,6  (+8h)
    u[4] = p[8];  u[5] = p[9];  u[6] = p[10]; u[7] = p[11];  // K = 16..22   (+8h)
    return __builtin_bit_cast(v16bf, u);
}

// Same fragment load but from LDS (bf16 h buffer) -> 2x ds_load_b128.
__device__ __forceinline__ v16bf load_tile_lds(const __bf16* base,
                                               int row, int ldk, int k0, int half) {
    const unsigned int* p =
        (const unsigned int*)(base + row * ldk + k0 + 8 * half);
    v8u u;
    u[0] = p[0];  u[1] = p[1];  u[2] = p[2];  u[3] = p[3];
    u[4] = p[8];  u[5] = p[9];  u[6] = p[10]; u[7] = p[11];
    return __builtin_bit_cast(v16bf, u);
}

// ---------------------------------------------------------------------------
// Elementwise conversion / gather kernels
// ---------------------------------------------------------------------------
__global__ void f32_to_bf16_kernel(const float* __restrict__ src,
                                   __bf16* __restrict__ dst, size_t n) {
    size_t i = (size_t)blockIdx.x * blockDim.x + threadIdx.x;
    if (i < n) dst[i] = (__bf16)src[i];
}

// pred_w is [K=HID][N=VOC]; produce predT [N][K] bf16 so K is contiguous.
__global__ void transpose_to_bf16_kernel(const float* __restrict__ src,
                                         __bf16* __restrict__ dst, int K, int N) {
    size_t i = (size_t)blockIdx.x * blockDim.x + threadIdx.x;
    if (i >= (size_t)K * N) return;
    int k = (int)(i / N);
    int n = (int)(i % N);
    dst[(size_t)n * K + k] = (__bf16)src[i];
}

// x_bf16[m][k] = emb_table[inp[m]][k]
__global__ void embed_gather_kernel(const int* __restrict__ inp,
                                    const float* __restrict__ emb,
                                    __bf16* __restrict__ x) {
    size_t i = (size_t)blockIdx.x * blockDim.x + threadIdx.x;
    if (i >= (size_t)M_ * EMB_) return;
    size_t m = i >> 9;            // / EMB_
    int    k = (int)(i & (EMB_ - 1));
    int idx = inp[m];
    x[i] = (__bf16)emb[(size_t)idx * EMB_ + k];
}

// hidden = z @ l2h_w + l2h_b, reshaped (torch .view) to [2][16][1024]
__global__ void latent2hidden_kernel(const float* __restrict__ z,
                                     const float* __restrict__ W,  // [256][2048]
                                     const float* __restrict__ bias,
                                     float* __restrict__ h0) {     // [2*16*1024]
    int i = blockIdx.x * blockDim.x + threadIdx.x;
    if (i >= 2 * B_ * HID_) return;
    int l  = i >> 14;
    int b2 = (i >> 10) & 15;
    int h  = i & 1023;
    int b  = l * 8 + (b2 >> 1);           // flat-reshape mapping
    int j  = ((b2 & 1) << 10) + h;
    float acc = bias[j];
    for (int k = 0; k < 256; ++k) acc += z[b * 256 + k] * W[k * 2048 + j];
    h0[i] = acc;
}

// ---------------------------------------------------------------------------
// Generic GEMM:  C[M][N] = A[M][K](bf16) * B[N][K](bf16)^T + bias0 + bias1
// Block = 256 threads (8 waves, 2x4), block tile 64(M) x 128(N);
// each wave: 32x32 C via 4 accumulators, K-step 32.
// Requires M%64==0, N%128==0, K%32==0 (true for all uses here).
// ---------------------------------------------------------------------------
__global__ void __launch_bounds__(256)
gemm_bf16_kernel(const __bf16* __restrict__ A, const __bf16* __restrict__ Bm,
                 const float* __restrict__ bias0, const float* __restrict__ bias1,
                 float* __restrict__ C, int M, int N, int K) {
    int wave = threadIdx.x >> 5;
    int lane = threadIdx.x & 31;
    int half = lane >> 4;
    int lc   = lane & 15;
    int m0 = blockIdx.y * 64 + (wave >> 2) * 32;
    int n0 = blockIdx.x * 128 + (wave & 3) * 32;

    v8f c00{}, c01{}, c10{}, c11{};
    for (int k0 = 0; k0 < K; k0 += 32) {
        v16bf a0 = load_tile(A, (size_t)(m0 + lc),      K, k0, half);
        v16bf a1 = load_tile(A, (size_t)(m0 + 16 + lc), K, k0, half);
        v16bf b0 = load_tile(Bm, (size_t)(n0 + lc),      K, k0, half);
        v16bf b1 = load_tile(Bm, (size_t)(n0 + 16 + lc), K, k0, half);
        c00 = wmma_bf16(a0, b0, c00);
        c01 = wmma_bf16(a0, b1, c01);
        c10 = wmma_bf16(a1, b0, c10);
        c11 = wmma_bf16(a1, b1, c11);
    }

    float biasA = 0.f, biasB = 0.f;
    if (bias0) { biasA += bias0[n0 + lc]; biasB += bias0[n0 + 16 + lc]; }
    if (bias1) { biasA += bias1[n0 + lc]; biasB += bias1[n0 + 16 + lc]; }

#pragma unroll
    for (int r = 0; r < 8; ++r) {
        int m = r + 8 * half;
        C[(size_t)(m0 + m) * N + n0 + lc]           = c00[r] + biasA;
        C[(size_t)(m0 + m) * N + n0 + 16 + lc]      = c01[r] + biasB;
        C[(size_t)(m0 + 16 + m) * N + n0 + lc]      = c10[r] + biasA;
        C[(size_t)(m0 + 16 + m) * N + n0 + 16 + lc] = c11[r] + biasB;
    }
}

// ---------------------------------------------------------------------------
// RNN recurrence for one layer, single 1024-thread workgroup (32 waves).
// h (16x1024) lives in LDS as bf16 (it is only consumed as a bf16 WMMA
// operand and as the bf16 hseq output, so no precision is lost vs f32 LDS,
// and the serial path avoids per-K-step converts: A fragment = 2x ds_load_b128).
// Per step: h_new = tanh(xw[:,t,:] + h @ Whh^T)  (f32 accumulate + f32 xw).
// ---------------------------------------------------------------------------
__global__ void __launch_bounds__(1024)
rnn_layer_kernel(const float* __restrict__ xw,    // [16][S][1024]
                 const float* __restrict__ h0,    // [16][1024] f32
                 const __bf16* __restrict__ Whh,  // [1024][1024] bf16, [n][k]
                 __bf16* __restrict__ hseq,       // [16][S][1024] bf16 out
                 int S) {
    __shared__ __bf16 hbuf[B_ * HID_];            // 32 KB

    int tid = threadIdx.x;
    for (int i = tid; i < B_ * HID_; i += 1024) hbuf[i] = (__bf16)h0[i];
    __syncthreads();

    int wave = tid >> 5, lane = tid & 31;
    int half = lane >> 4, lc = lane & 15;
    int n0 = wave * 32;

    for (int t = 0; t < S; ++t) {
        v8f c0{}, c1{};
        for (int k0 = 0; k0 < HID_; k0 += 32) {
            v16bf a  = load_tile_lds(hbuf, lc, HID_, k0, half);   // rows = batch
            v16bf b0 = load_tile(Whh, (size_t)(n0 + lc),      HID_, k0, half);
            v16bf b1 = load_tile(Whh, (size_t)(n0 + 16 + lc), HID_, k0, half);
            c0 = wmma_bf16(a, b0, c0);
            c1 = wmma_bf16(a, b1, c1);
        }

        __bf16 hn0[8], hn1[8];
#pragma unroll
        for (int r = 0; r < 8; ++r) {
            int m = r + 8 * half;                       // batch index
            size_t off = ((size_t)m * S + t) * HID_;
            hn0[r] = (__bf16)tanhf(xw[off + n0 + lc]      + c0[r]);
            hn1[r] = (__bf16)tanhf(xw[off + n0 + 16 + lc] + c1[r]);
        }

        __syncthreads();   // everyone done reading hbuf for this step
#pragma unroll
        for (int r = 0; r < 8; ++r) {
            int m = r + 8 * half;
            hbuf[m * HID_ + n0 + lc]      = hn0[r];
            hbuf[m * HID_ + n0 + 16 + lc] = hn1[r];
            size_t off = ((size_t)m * S + t) * HID_;
            hseq[off + n0 + lc]      = hn0[r];
            hseq[off + n0 + 16 + lc] = hn1[r];
        }
        __syncthreads();   // hbuf fully updated before next step's reads
    }
}

// ---------------------------------------------------------------------------
// Host-side orchestration
// ---------------------------------------------------------------------------
extern "C" void kernel_launch(void* const* d_in, const int* in_sizes, int n_in,
                              void* d_out, int out_size, void* d_ws, size_t ws_size,
                              hipStream_t stream) {
    const float* z      = (const float*)d_in[0];
    const int*   inp    = (const int*)  d_in[1];
    const float* emb    = (const float*)d_in[2];
    const float* l2h_w  = (const float*)d_in[3];
    const float* l2h_b  = (const float*)d_in[4];
    const float* w_ih0  = (const float*)d_in[5];
    const float* w_hh0  = (const float*)d_in[6];
    const float* b_ih0  = (const float*)d_in[7];
    const float* b_hh0  = (const float*)d_in[8];
    const float* w_ih1  = (const float*)d_in[9];
    const float* w_hh1  = (const float*)d_in[10];
    const float* b_ih1  = (const float*)d_in[11];
    const float* b_hh1  = (const float*)d_in[12];
    const float* pred_w = (const float*)d_in[13];
    const float* pred_b = (const float*)d_in[14];
    float* out = (float*)d_out;

    // ---- workspace layout (all sizes 256B-aligned) ----
    char* w = (char*)d_ws;
    __bf16* w_ih0b = (__bf16*)w;  w += (size_t)HID_ * EMB_ * 2;   // 1 MiB
    __bf16* w_hh0b = (__bf16*)w;  w += (size_t)HID_ * HID_ * 2;   // 2 MiB
    __bf16* w_ih1b = (__bf16*)w;  w += (size_t)HID_ * HID_ * 2;
    __bf16* w_hh1b = (__bf16*)w;  w += (size_t)HID_ * HID_ * 2;
    __bf16* predTb = (__bf16*)w;  w += (size_t)VOC_ * HID_ * 2;   // 62.5 MiB
    float*  h0     = (float*)w;   w += (size_t)2 * B_ * HID_ * 4; // 128 KiB
    __bf16* h2b    = (__bf16*)w;  w += (size_t)M_ * HID_ * 2;     // 16 MiB

    // ---- large transients live in d_out (fully overwritten by final GEMM) ----
    char* ob = (char*)d_out;
    __bf16* xb  = (__bf16*)(ob);                       //  8 MiB: x gathered, bf16
    float*  xw  = (float*) (ob + ((size_t)64  << 20)); // 32 MiB: xw0 then xw1
    __bf16* h1b = (__bf16*)(ob + ((size_t)256 << 20)); // 16 MiB: layer-0 output

    // 1) weight conversions to bf16
    {
        size_t n;
        n = (size_t)HID_ * EMB_;
        f32_to_bf16_kernel<<<(n + 255) / 256, 256, 0, stream>>>(w_ih0, w_ih0b, n);
        n = (size_t)HID_ * HID_;
        f32_to_bf16_kernel<<<(n + 255) / 256, 256, 0, stream>>>(w_hh0, w_hh0b, n);
        f32_to_bf16_kernel<<<(n + 255) / 256, 256, 0, stream>>>(w_ih1, w_ih1b, n);
        f32_to_bf16_kernel<<<(n + 255) / 256, 256, 0, stream>>>(w_hh1, w_hh1b, n);
        n = (size_t)HID_ * VOC_;
        transpose_to_bf16_kernel<<<(n + 255) / 256, 256, 0, stream>>>(pred_w, predTb,
                                                                      HID_, VOC_);
    }

    // 2) embedding gather -> bf16
    {
        size_t n = (size_t)M_ * EMB_;
        embed_gather_kernel<<<(n + 255) / 256, 256, 0, stream>>>(inp, emb, xb);
    }

    // 3) initial hidden states [2][16][1024]
    latent2hidden_kernel<<<(2 * B_ * HID_ + 255) / 256, 256, 0, stream>>>(
        z, l2h_w, l2h_b, h0);

    // 4) xw0 = x @ w_ih0^T + b_ih0 + b_hh0     (M=8192, N=1024, K=512)
    gemm_bf16_kernel<<<dim3(HID_ / 128, M_ / 64), 256, 0, stream>>>(
        xb, w_ih0b, b_ih0, b_hh0, xw, M_, HID_, EMB_);

    // 5) layer-0 recurrence -> h1 (bf16)
    rnn_layer_kernel<<<1, 1024, 0, stream>>>(xw, h0, w_hh0b, h1b, S_);

    // 6) xw1 = h1 @ w_ih1^T + b_ih1 + b_hh1    (M=8192, N=1024, K=1024)
    gemm_bf16_kernel<<<dim3(HID_ / 128, M_ / 64), 256, 0, stream>>>(
        h1b, w_ih1b, b_ih1, b_hh1, xw, M_, HID_, HID_);

    // 7) layer-1 recurrence -> h2 (bf16, in ws: stays alive while out is written)
    rnn_layer_kernel<<<1, 1024, 0, stream>>>(xw, h0 + B_ * HID_, w_hh1b, h2b, S_);

    // 8) logits = h2 @ pred_w + pred_b         (M=8192, N=32000, K=1024)
    gemm_bf16_kernel<<<dim3(VOC_ / 128, M_ / 64), 256, 0, stream>>>(
        h2b, predTb, pred_b, nullptr, out, M_, VOC_, HID_);
}
